// CoattentiontraceptionBlock_128849018989
// MI455X (gfx1250) — matrix operand order repaired and checked by reading, not compiled, not run on hardware
//
#include <hip/hip_runtime.h>

// ---------------- problem constants ----------------
constexpr int cB = 16, cLV = 1026, cLQ = 100, cHID = 1280, cH = 8, cDH = 160;
constexpr int cHPG = 2;                       // heads per conv group
constexpr long long SZX = (long long)cB * cLV * cHID;   // 21,012,480
constexpr long long SZY = (long long)cB * cLQ * cHID;   //  2,048,000
constexpr long long SZS = (long long)cB * cH * cLV * cLQ; // 13,132,800
constexpr float RSQRT_DH = 0.07905694150420949f;        // 1/sqrt(160)

typedef __attribute__((ext_vector_type(16))) __bf16 v16bf;
typedef __attribute__((ext_vector_type(8)))  __bf16 v8bf;
typedef __attribute__((ext_vector_type(8)))  float  v8f;

#define F_NT    1   // B matrix is [N,K] row-major (weights); else [K,N]
#define F_RELU  2
#define F_SBIAS 8   // bias is a scalar (size-1), broadcast

#define AS1 __attribute__((address_space(1)))
#define AS3 __attribute__((address_space(3)))

// vector pointee types matching the async-to-LDS builtin prototypes
typedef int v4i_ __attribute__((vector_size(16)));
typedef int v2i_ __attribute__((vector_size(8)));

#if defined(__gfx1250__) && \
    __has_builtin(__builtin_amdgcn_global_load_async_to_lds_b128) && \
    __has_builtin(__builtin_amdgcn_global_load_async_to_lds_b64)
#define HAS_ASYNC_LDS 1
#else
#define HAS_ASYNC_LDS 0
#endif

#if HAS_ASYNC_LDS
#if __has_builtin(__builtin_amdgcn_s_wait_asynccnt)
#define WAIT_ASYNC() __builtin_amdgcn_s_wait_asynccnt(0)
#else
#define WAIT_ASYNC() asm volatile("s_wait_asynccnt 0" ::: "memory")
#endif
#endif

// f32x4 quads -> bf16x16 fragment
__device__ inline v16bf cvt_frag(float4 a0, float4 a1, float4 a2, float4 a3)
{
  v16bf o;
  o[0]=(__bf16)a0.x;  o[1]=(__bf16)a0.y;  o[2]=(__bf16)a0.z;  o[3]=(__bf16)a0.w;
  o[4]=(__bf16)a1.x;  o[5]=(__bf16)a1.y;  o[6]=(__bf16)a1.z;  o[7]=(__bf16)a1.w;
  o[8]=(__bf16)a2.x;  o[9]=(__bf16)a2.y;  o[10]=(__bf16)a2.z; o[11]=(__bf16)a2.w;
  o[12]=(__bf16)a3.x; o[13]=(__bf16)a3.y; o[14]=(__bf16)a3.z; o[15]=(__bf16)a3.w;
  return o;
}

// =====================================================================
// Generic batched GEMM:  C[z] = relu?( alpha * A[z] @ op(B[z]) + bias (+res) )
// A: f32 [M,K] (lda); B: f32 NT [N,K] or NN [K,N]. Tiles staged to LDS as f32
// (via GLOBAL_LOAD_ASYNC_TO_LDS when available), converted to bf16 at
// fragment read, consumed by v_wmma_f32_16x16x32_bf16.
// Block tile 128x64 (8 waves, each 32x32 via 2x2 WMMA).
// =====================================================================
__global__ __launch_bounds__(256) void gemm_wmma_kernel(
    const float* __restrict__ A, const float* __restrict__ Bm,
    const float* __restrict__ bias, const float* __restrict__ res,
    float* __restrict__ C,
    int M, int N, int K, int lda, int ldb, int ldc, int hdiv,
    long long sA1, long long sA2, long long sB1, long long sB2,
    long long sC1, long long sC2, float alpha, int flags)
{
  __shared__ float AsF[128][32];   // 16 KB
  __shared__ float BsF[64][32];    //  8 KB

  const int z  = blockIdx.z;
  const int bz = z / hdiv;
  const int hz = z - bz * hdiv;
  const float* Ab = A  + bz * sA1 + hz * sA2;
  const float* Bb = Bm + bz * sB1 + hz * sB2;
  float*       Cb = C  + bz * sC1 + hz * sC2;
  const float* Rb = res ? res + bz * sC1 + hz * sC2 : (const float*)0;

  const int m0   = blockIdx.y * 128;
  const int n0   = blockIdx.x * 64;
  const int t    = threadIdx.x;
  const int lane = t & 31;
  const int wave = t >> 5;        // 0..7
  const int wm   = wave >> 1;     // 0..3 -> 32-row strip
  const int wn   = wave & 1;      // 0..1 -> 32-col strip
  const int r    = lane & 15;
  const int hi   = lane >> 4;
  const bool bNT = (flags & F_NT) != 0;
  const bool mnFull = (m0 + 128 <= M) && (n0 + 64 <= N);

  // per-thread staging coordinates
  const int arow = t >> 1;          // A: 0..127
  const int acol = (t & 1) * 16;    // A: 0 or 16
  const int brow = t >> 2;          // B: 0..63
  const int bcol = (t & 3) * 8;     // B: 0,8,16,24

  v8f acc[2][2] = {};

  for (int kb = 0; kb < K; kb += 32) {
    const bool full = mnFull && (kb + 32 <= K);   // block-uniform
    if (full) {
      { // ---- A fast path: 16 consecutive f32 per thread ----
        const float* asrc = Ab + (long long)(m0 + arow) * lda + kb + acol;
        float* adst = &AsF[arow][acol];
#if HAS_ASYNC_LDS
        if ((lda & 3) == 0) {
#pragma unroll
          for (int i = 0; i < 4; ++i)
            __builtin_amdgcn_global_load_async_to_lds_b128(
                (AS1 v4i_*)(asrc + 4 * i), (AS3 v4i_*)(adst + 4 * i), 0, 0);
        } else {  // rows only 8B aligned (e.g. lda == 1026)
#pragma unroll
          for (int i = 0; i < 8; ++i)
            __builtin_amdgcn_global_load_async_to_lds_b64(
                (AS1 v2i_*)(asrc + 2 * i), (AS3 v2i_*)(adst + 2 * i), 0, 0);
        }
#else
        const float2* s2 = (const float2*)asrc;
        float2 f[8];
#pragma unroll
        for (int i = 0; i < 8; ++i) f[i] = s2[i];
#pragma unroll
        for (int i = 0; i < 4; ++i) {
          float4 v; v.x = f[2*i].x; v.y = f[2*i].y; v.z = f[2*i+1].x; v.w = f[2*i+1].y;
          *(float4*)(adst + 4 * i) = v;
        }
#endif
      }
      if (bNT) { // ---- B fast path: 8 consecutive f32 per thread ----
        const float* bsrc = Bb + (long long)(n0 + brow) * ldb + kb + bcol;
        float* bdst = &BsF[brow][bcol];
#if HAS_ASYNC_LDS
        if ((ldb & 3) == 0) {
#pragma unroll
          for (int i = 0; i < 2; ++i)
            __builtin_amdgcn_global_load_async_to_lds_b128(
                (AS1 v4i_*)(bsrc + 4 * i), (AS3 v4i_*)(bdst + 4 * i), 0, 0);
        } else {
#pragma unroll
          for (int i = 0; i < 4; ++i)
            __builtin_amdgcn_global_load_async_to_lds_b64(
                (AS1 v2i_*)(bsrc + 2 * i), (AS3 v2i_*)(bdst + 2 * i), 0, 0);
        }
#else
        const float2* s2 = (const float2*)bsrc;
        float2 f[4];
#pragma unroll
        for (int i = 0; i < 4; ++i) f[i] = s2[i];
#pragma unroll
        for (int i = 0; i < 2; ++i) {
          float4 v; v.x = f[2*i].x; v.y = f[2*i].y; v.z = f[2*i+1].x; v.w = f[2*i+1].y;
          *(float4*)(bdst + 4 * i) = v;
        }
#endif
      } else {   // ---- B NN: column-strided gather through registers ----
        const float* bsrc = Bb + (long long)(kb + bcol) * ldb + (n0 + brow);
        float f[8];
#pragma unroll
        for (int i = 0; i < 8; ++i) f[i] = bsrc[(long long)i * ldb];
        float4 v0; v0.x = f[0]; v0.y = f[1]; v0.z = f[2]; v0.w = f[3];
        float4 v1; v1.x = f[4]; v1.y = f[5]; v1.z = f[6]; v1.w = f[7];
        *(float4*)(&BsF[brow][bcol])     = v0;
        *(float4*)(&BsF[brow][bcol + 4]) = v1;
      }
#if HAS_ASYNC_LDS
      WAIT_ASYNC();
#endif
    } else {
      { // ---- A guarded path ----
        const int gm = m0 + arow;
#pragma unroll
        for (int i = 0; i < 16; ++i) {
          const int gk = kb + acol + i;
          AsF[arow][acol + i] =
              (gm < M && gk < K) ? Ab[(long long)gm * lda + gk] : 0.0f;
        }
      }
      { // ---- B guarded path ----
        const int gn = n0 + brow;
#pragma unroll
        for (int i = 0; i < 8; ++i) {
          const int gk = kb + bcol + i;
          float v = 0.0f;
          if (gn < N && gk < K)
            v = bNT ? Bb[(long long)gn * ldb + gk]
                    : Bb[(long long)gk * ldb + gn];
          BsF[brow][bcol + i] = v;
        }
      }
    }
    __syncthreads();

    // A fragments: lane r = row M, K halves {hi*8..+7} U {16+hi*8..+7}
    v16bf af[2], bfr[2];
#pragma unroll
    for (int tm = 0; tm < 2; ++tm) {
      const float* rp = &AsF[wm * 32 + tm * 16 + r][0];
      float4 a0 = *(const float4*)(rp + hi * 8);
      float4 a1 = *(const float4*)(rp + hi * 8 + 4);
      float4 a2 = *(const float4*)(rp + 16 + hi * 8);
      float4 a3 = *(const float4*)(rp + 16 + hi * 8 + 4);
      af[tm] = cvt_frag(a0, a1, a2, a3);
    }
    // B fragments: lane r = col N, K halves {hi*8..+7} U {16+hi*8..+7}
#pragma unroll
    for (int tn = 0; tn < 2; ++tn) {
      const float* rp = &BsF[wn * 32 + tn * 16 + r][hi * 8];
      float4 b0 = *(const float4*)(rp);
      float4 b1 = *(const float4*)(rp + 4);
      float4 b2 = *(const float4*)(rp + 16);
      float4 b3 = *(const float4*)(rp + 20);
      bfr[tn] = cvt_frag(b0, b1, b2, b3);
    }
#pragma unroll
    for (int tm = 0; tm < 2; ++tm)
#pragma unroll
      for (int tn = 0; tn < 2; ++tn)
        acc[tm][tn] = __builtin_amdgcn_wmma_f32_16x16x32_bf16(
            false, af[tm], false, bfr[tn], (short)0, acc[tm][tn], false, false);
    __syncthreads();
  }

  // epilogue: D layout lane r = N, VGPR v -> M = hi*8 + v
#pragma unroll
  for (int tm = 0; tm < 2; ++tm)
#pragma unroll
    for (int tn = 0; tn < 2; ++tn) {
      const int cn = n0 + wn * 32 + tn * 16 + r;
      if (cn >= N) continue;
      const float bv = bias ? ((flags & F_SBIAS) ? bias[0] : bias[cn]) : 0.0f;
#pragma unroll
      for (int v = 0; v < 8; ++v) {
        const int cm = m0 + wm * 32 + tm * 16 + hi * 8 + v;
        if (cm >= M) continue;
        float x = acc[tm][tn][v] * alpha + bv;
        if (Rb) x += Rb[(long long)cm * ldc + cn];
        if (flags & F_RELU) x = fmaxf(x, 0.0f);
        Cb[(long long)cm * ldc + cn] = x;
      }
    }
}

// ---------------- masks: mask[row] = (sum |x| == 0) ----------------
__global__ __launch_bounds__(256) void mask_kernel(const float* __restrict__ X,
                                                   float* __restrict__ mask, int D)
{
  const float* x = X + (long long)blockIdx.x * D;
  __shared__ float red[256];
  const int t = threadIdx.x;
  float s = 0.f;
  for (int c = t; c < D; c += 256) s += fabsf(x[c]);
  red[t] = s; __syncthreads();
  for (int o = 128; o > 0; o >>= 1) { if (t < o) red[t] += red[t + o]; __syncthreads(); }
  if (t == 0) mask[blockIdx.x] = (red[0] == 0.0f) ? 1.0f : 0.0f;
}

// ---------------- causal depthwise conv over head groups ----------------
// in/out: [B, S, HID]; group 0 (cols 0..319) pass-through; groups 1..3 use k=3,5,7
__global__ __launch_bounds__(256) void conv_kernel(
    const float* __restrict__ in, float* __restrict__ out,
    const float* __restrict__ w3, const float* __restrict__ b3,
    const float* __restrict__ w5, const float* __restrict__ b5,
    const float* __restrict__ w7, const float* __restrict__ b7, int S)
{
  const long long total = (long long)cB * S * cHID;
  const long long idx = (long long)blockIdx.x * 256 + threadIdx.x;
  if (idx >= total) return;
  const int col = (int)(idx % cHID);
  const long long bs = idx / cHID;
  const int s  = (int)(bs % S);
  const int bb = (int)(bs / S);
  const int grp = col / (cHPG * cDH);
  if (grp == 0) { out[idx] = in[idx]; return; }
  const int d = col % cDH;
  const float* w; const float* bi; int k;
  if (grp == 1)      { w = w3; bi = b3; k = 3; }
  else if (grp == 2) { w = w5; bi = b5; k = 5; }
  else               { w = w7; bi = b7; k = 7; }
  float acc = bi[d];
  for (int i = 0; i < k; ++i) {
    const int sp = s - (k - 1) + i;
    if (sp >= 0) acc += in[((long long)bb * S + sp) * cHID + col] * w[d * k + i];
  }
  out[idx] = acc;
}

// ---------------- masked softmax over last dim (in place) ----------------
// S: [batch][rows][cols]; colmask[b*maskld+c] masks keys; rowmask masks whole row.
__global__ __launch_bounds__(256) void softmax_kernel(
    float* __restrict__ S, const float* __restrict__ colmask,
    const float* __restrict__ rowmask,
    int rows, int cols, int hdiv, int maskld, int rowmaskld)
{
  const int z   = blockIdx.y;
  const int b   = z / hdiv;
  const int row = blockIdx.x;
  float* Sr = S + ((long long)z * rows + row) * cols;
  const int t = threadIdx.x;
  __shared__ float red[256];
  const bool rowMasked = rowmask && (rowmask[(long long)b * rowmaskld + row] != 0.0f);

  float mx = -3.4e38f;
  for (int c = t; c < cols; c += 256) {
    float v = Sr[c];
    if (colmask && colmask[(long long)b * maskld + c] != 0.0f) v = -10000.0f;
    if (rowMasked) v = -10000.0f;
    Sr[c] = v;
    mx = fmaxf(mx, v);
  }
  red[t] = mx; __syncthreads();
  for (int o = 128; o > 0; o >>= 1) { if (t < o) red[t] = fmaxf(red[t], red[t + o]); __syncthreads(); }
  mx = red[0]; __syncthreads();

  float sum = 0.f;
  for (int c = t; c < cols; c += 256) { float e = __expf(Sr[c] - mx); Sr[c] = e; sum += e; }
  red[t] = sum; __syncthreads();
  for (int o = 128; o > 0; o >>= 1) { if (t < o) red[t] += red[t + o]; __syncthreads(); }
  const float inv = 1.0f / red[0];
  for (int c = t; c < cols; c += 256) Sr[c] *= inv;
}

// ---------------- layernorm in place; mode 0 = custom_ln, 1 = torch_ln ----
__global__ __launch_bounds__(256) void ln_kernel(float* __restrict__ X,
    const float* __restrict__ g, const float* __restrict__ b, int D, int mode)
{
  float* x = X + (long long)blockIdx.x * D;
  __shared__ float red[256];
  const int t = threadIdx.x;
  float s = 0.f;
  for (int c = t; c < D; c += 256) s += x[c];
  red[t] = s; __syncthreads();
  for (int o = 128; o > 0; o >>= 1) { if (t < o) red[t] += red[t + o]; __syncthreads(); }
  const float mean = red[0] / D; __syncthreads();
  float s2 = 0.f;
  for (int c = t; c < D; c += 256) { float d = x[c] - mean; s2 += d * d; }
  red[t] = s2; __syncthreads();
  for (int o = 128; o > 0; o >>= 1) { if (t < o) red[t] += red[t + o]; __syncthreads(); }
  const float vs = red[0];
  if (mode == 0) {
    const float inv = 1.0f / (sqrtf(vs / (D - 1)) + 1e-6f);
    for (int c = t; c < D; c += 256) x[c] = g[c] * (x[c] - mean) * inv + b[c];
  } else {
    const float inv = rsqrtf(vs / D + 1e-5f);
    for (int c = t; c < D; c += 256) x[c] = (x[c] - mean) * inv * g[c] + b[c];
  }
}

// ---------------- elementwise helpers ----------------
__global__ __launch_bounds__(256) void scale_cols_kernel(
    const float* __restrict__ in, const float* __restrict__ aw,
    float* __restrict__ out, long long n)
{
  const long long i = (long long)blockIdx.x * 256 + threadIdx.x;
  if (i < n) out[i] = in[i] * aw[i % cHID];
}

__global__ __launch_bounds__(256) void fuse_kernel(
    const float* __restrict__ x, const float* __restrict__ pool,
    const float* __restrict__ alpha, float* __restrict__ out, long long n)
{
  const long long i = (long long)blockIdx.x * 256 + threadIdx.x;
  if (i < n) out[i] = alpha[0] * x[i] * pool[i] + x[i];
}

// ---------------- fc3 on sliced rows [:,1:-1,:] ----------------
__global__ __launch_bounds__(256) void fc3_kernel(
    const float* __restrict__ h2, const float* __restrict__ w,
    const float* __restrict__ b, float* __restrict__ out)
{
  const int total = cB * (cLV - 2) * 2;
  const int idx = blockIdx.x * 256 + threadIdx.x;
  if (idx >= total) return;
  const int o = idx & 1;
  const int vs = idx >> 1;
  const int s  = vs % (cLV - 2);
  const int bb = vs / (cLV - 2);
  const float* hr = h2 + ((long long)bb * cLV + (s + 1)) * 320;
  float acc = b[o];
  for (int i = 0; i < 320; ++i) acc += hr[i] * w[o * 320 + i];
  out[idx] = acc;
}

// =====================================================================
// Host side
// =====================================================================
struct Params {
  const float *V, *Q;
  const float *xy_qw,*xy_qb,*xy_kw,*xy_kb,*xy_vw,*xy_vb,*xy_mw,*xy_mb;
  const float *xy_cw[3],*xy_cb[3];
  const float *yx_qw,*yx_qb,*yx_kw,*yx_kb,*yx_vw,*yx_vb,*yx_mw,*yx_mb;
  const float *yx_ckw[3],*yx_ckb[3],*yx_cvw[3],*yx_cvb[3];
  const float *fx_l1w,*fx_l1b,*fx_l2w,*fx_l2b;
  const float *fy_l1w,*fy_l1b,*fy_l2w,*fy_l2b;
  const float *n2x_g,*n2x_b,*n2y_g,*n2y_b,*n3x_g,*n3x_b,*n3y_g,*n3y_b;
  const float *att_w,*att_b;
  const float *fc1w,*fc1b,*ln1g,*ln1b,*fc2w,*fc2b,*ln2g,*ln2b,*fc3w,*fc3b;
  const float *alpha;
};

static inline void gemm(hipStream_t s, const float* A, const float* B,
                        const float* bias, const float* res, float* C,
                        int M, int N, int K, int lda, int ldb, int ldc,
                        int batch, int hdiv,
                        long long sA1, long long sA2, long long sB1, long long sB2,
                        long long sC1, long long sC2, float alpha, int flags)
{
  dim3 g((N + 63) / 64, (M + 127) / 128, batch);
  gemm_wmma_kernel<<<g, 256, 0, s>>>(A, B, bias, res, C, M, N, K, lda, ldb, ldc,
                                     hdiv, sA1, sA2, sB1, sB2, sC1, sC2, alpha, flags);
}

static inline unsigned eg(long long n) { return (unsigned)((n + 255) / 256); }

extern "C" void kernel_launch(void* const* d_in, const int* in_sizes, int n_in,
                              void* d_out, int out_size, void* d_ws, size_t ws_size,
                              hipStream_t stream)
{
  (void)out_size; (void)ws_size;
  Params p;
  auto F = [&](int i) { return (const float*)d_in[i]; };
  const bool sorted = (n_in > 0 && in_sizes[0] == 1);
  int i = 0;
  if (!sorted) {
    // insertion order: v, q, then params in construction order
    p.V = F(i++); p.Q = F(i++);
    p.xy_qw=F(i++); p.xy_qb=F(i++); p.xy_kw=F(i++); p.xy_kb=F(i++);
    p.xy_vw=F(i++); p.xy_vb=F(i++); p.xy_mw=F(i++); p.xy_mb=F(i++);
    for (int k=0;k<3;++k){ p.xy_cw[k]=F(i++); p.xy_cb[k]=F(i++); }
    p.yx_qw=F(i++); p.yx_qb=F(i++); p.yx_kw=F(i++); p.yx_kb=F(i++);
    p.yx_vw=F(i++); p.yx_vb=F(i++); p.yx_mw=F(i++); p.yx_mb=F(i++);
    for (int k=0;k<3;++k){ p.yx_ckw[k]=F(i++); p.yx_ckb[k]=F(i++); }
    for (int k=0;k<3;++k){ p.yx_cvw[k]=F(i++); p.yx_cvb[k]=F(i++); }
    p.fx_l1w=F(i++); p.fx_l1b=F(i++); p.fx_l2w=F(i++); p.fx_l2b=F(i++);
    p.fy_l1w=F(i++); p.fy_l1b=F(i++); p.fy_l2w=F(i++); p.fy_l2b=F(i++);
    p.n2x_g=F(i++); p.n2x_b=F(i++); p.n2y_g=F(i++); p.n2y_b=F(i++);
    p.n3x_g=F(i++); p.n3x_b=F(i++); p.n3y_g=F(i++); p.n3y_b=F(i++);
    p.att_w=F(i++); p.att_b=F(i++);
    p.fc1w=F(i++); p.fc1b=F(i++); p.ln1g=F(i++); p.ln1b=F(i++);
    p.fc2w=F(i++); p.fc2b=F(i++); p.ln2g=F(i++); p.ln2b=F(i++);
    p.fc3w=F(i++); p.fc3b=F(i++);
    p.alpha=F(i++);
  } else {
    // jax pytree sorted-key order: params (sorted), q, v; dicts by sorted key
    p.alpha=F(i++);
    p.att_b=F(i++); p.att_w=F(i++);
    p.fc1b=F(i++); p.fc1w=F(i++);
    p.fc2b=F(i++); p.fc2w=F(i++);
    p.fc3b=F(i++); p.fc3w=F(i++);
    p.fx_l1b=F(i++); p.fx_l1w=F(i++); p.fx_l2b=F(i++); p.fx_l2w=F(i++);
    p.fy_l1b=F(i++); p.fy_l1w=F(i++); p.fy_l2b=F(i++); p.fy_l2w=F(i++);
    p.ln1b=F(i++); p.ln1g=F(i++); p.ln2b=F(i++); p.ln2g=F(i++);
    p.n2x_b=F(i++); p.n2x_g=F(i++); p.n2y_b=F(i++); p.n2y_g=F(i++);
    p.n3x_b=F(i++); p.n3x_g=F(i++); p.n3y_b=F(i++); p.n3y_g=F(i++);
    for (int k=0;k<3;++k){ p.xy_cb[k]=F(i++); p.xy_cw[k]=F(i++); }
    p.xy_kb=F(i++); p.xy_kw=F(i++); p.xy_mb=F(i++); p.xy_mw=F(i++);
    p.xy_qb=F(i++); p.xy_qw=F(i++); p.xy_vb=F(i++); p.xy_vw=F(i++);
    for (int k=0;k<3;++k){ p.yx_ckb[k]=F(i++); p.yx_ckw[k]=F(i++); }
    for (int k=0;k<3;++k){ p.yx_cvb[k]=F(i++); p.yx_cvw[k]=F(i++); }
    p.yx_kb=F(i++); p.yx_kw=F(i++); p.yx_mb=F(i++); p.yx_mw=F(i++);
    p.yx_qb=F(i++); p.yx_qw=F(i++); p.yx_vb=F(i++); p.yx_vw=F(i++);
    p.Q=F(i++); p.V=F(i++);
  }

  // ---------------- workspace carve ----------------
  float* w = (float*)d_ws;
  float* bufA = w; w += SZX;
  float* bufB = w; w += SZX;
  float* bufC = w; w += SZX;
  float* bufD = w; w += SZX;
  float* bufS = w; w += SZS;
  float* yb1  = w; w += SZY;
  float* yb2  = w; w += SZY;
  float* yb3  = w; w += SZY;
  float* yb4  = w; w += SZY;
  float* mv   = w; w += cB * cLV;
  float* mq   = w; w += cB * cLQ;

  const int MX = cB * cLV;   // 16416
  const int MY = cB * cLQ;   // 1600
  const long long XH = (long long)cLV * cHID, YH = (long long)cLQ * cHID;

  // masks
  mask_kernel<<<MX, 256, 0, stream>>>(p.V, mv, cHID);
  mask_kernel<<<MY, 256, 0, stream>>>(p.Q, mq, cHID);

  // ---- xy cross-attention: q from x(=v), k/v from y(=q), conv on q heads ----
  gemm(stream, p.V, p.xy_qw, p.xy_qb, nullptr, bufA, MX, cHID, cHID, cHID, cHID, cHID,
       1, 1, 0,0,0,0,0,0, 1.f, F_NT);
  conv_kernel<<<eg(SZX), 256, 0, stream>>>(bufA, bufB, p.xy_cw[0],p.xy_cb[0],
       p.xy_cw[1],p.xy_cb[1], p.xy_cw[2],p.xy_cb[2], cLV);
  gemm(stream, p.Q, p.xy_kw, p.xy_kb, nullptr, yb1, MY, cHID, cHID, cHID, cHID, cHID,
       1, 1, 0,0,0,0,0,0, 1.f, F_NT);
  gemm(stream, p.Q, p.xy_vw, p.xy_vb, nullptr, yb2, MY, cHID, cHID, cHID, cHID, cHID,
       1, 1, 0,0,0,0,0,0, 1.f, F_NT);
  // scores [B,H,LV,LQ] = qconv . k^T / sqrt(DH)
  gemm(stream, bufB, yb1, nullptr, nullptr, bufS, cLV, cLQ, cDH, cHID, cHID, cLQ,
       cB*cH, cH, XH, cDH, YH, cDH, (long long)cH*cLV*cLQ, (long long)cLV*cLQ,
       RSQRT_DH, F_NT);
  softmax_kernel<<<dim3(cLV, cB*cH), 256, 0, stream>>>(bufS, mq, nullptr,
       cLV, cLQ, cH, cLQ, 0);
  // ctx = att @ v  -> bufA at head offsets
  gemm(stream, bufS, yb2, nullptr, nullptr, bufA, cLV, cDH, cLQ, cLQ, cHID, cHID,
       cB*cH, cH, (long long)cH*cLV*cLQ, (long long)cLV*cLQ, YH, cDH, XH, cDH, 1.f, 0);
  // x = custom_ln( v + m(ctx) )
  gemm(stream, bufA, p.xy_mw, p.xy_mb, p.V, bufB, MX, cHID, cHID, cHID, cHID, cHID,
       1, 1, 0,0,0,0,0,0, 1.f, F_NT);
  ln_kernel<<<MX, 256, 0, stream>>>(bufB, p.n2x_g, p.n2x_b, cHID, 0);   // x = bufB

  // ---- yx cross-attention: q from y(=q), k/v from updated x, conv on k/v ----
  gemm(stream, bufB, p.yx_kw, p.yx_kb, nullptr, bufA, MX, cHID, cHID, cHID, cHID, cHID,
       1, 1, 0,0,0,0,0,0, 1.f, F_NT);
  conv_kernel<<<eg(SZX), 256, 0, stream>>>(bufA, bufC, p.yx_ckw[0],p.yx_ckb[0],
       p.yx_ckw[1],p.yx_ckb[1], p.yx_ckw[2],p.yx_ckb[2], cLV);
  gemm(stream, bufB, p.yx_vw, p.yx_vb, nullptr, bufA, MX, cHID, cHID, cHID, cHID, cHID,
       1, 1, 0,0,0,0,0,0, 1.f, F_NT);
  conv_kernel<<<eg(SZX), 256, 0, stream>>>(bufA, bufD, p.yx_cvw[0],p.yx_cvb[0],
       p.yx_cvw[1],p.yx_cvb[1], p.yx_cvw[2],p.yx_cvb[2], cLV);
  gemm(stream, p.Q, p.yx_qw, p.yx_qb, nullptr, yb1, MY, cHID, cHID, cHID, cHID, cHID,
       1, 1, 0,0,0,0,0,0, 1.f, F_NT);
  // scores [B,H,LQ,LV]
  gemm(stream, yb1, bufC, nullptr, nullptr, bufS, cLQ, cLV, cDH, cHID, cHID, cLV,
       cB*cH, cH, YH, cDH, XH, cDH, (long long)cH*cLQ*cLV, (long long)cLQ*cLV,
       RSQRT_DH, F_NT);
  softmax_kernel<<<dim3(cLQ, cB*cH), 256, 0, stream>>>(bufS, mv, nullptr,
       cLQ, cLV, cH, cLV, 0);
  gemm(stream, bufS, bufD, nullptr, nullptr, yb2, cLQ, cDH, cLV, cLV, cHID, cHID,
       cB*cH, cH, (long long)cH*cLQ*cLV, (long long)cLQ*cLV, XH, cDH, YH, cDH, 1.f, 0);
  gemm(stream, yb2, p.yx_mw, p.yx_mb, p.Q, yb3, MY, cHID, cHID, cHID, cHID, cHID,
       1, 1, 0,0,0,0,0,0, 1.f, F_NT);
  ln_kernel<<<MY, 256, 0, stream>>>(yb3, p.n2y_g, p.n2y_b, cHID, 0);    // y = yb3

  // ---- FFN x ----
  gemm(stream, bufB, p.fx_l1w, p.fx_l1b, nullptr, bufA, MX, cHID, cHID, cHID, cHID, cHID,
       1, 1, 0,0,0,0,0,0, 1.f, F_NT | F_RELU);
  gemm(stream, bufA, p.fx_l2w, p.fx_l2b, bufB, bufC, MX, cHID, cHID, cHID, cHID, cHID,
       1, 1, 0,0,0,0,0,0, 1.f, F_NT);
  ln_kernel<<<MX, 256, 0, stream>>>(bufC, p.n3x_g, p.n3x_b, cHID, 0);   // x = bufC
  // ---- FFN y ----
  gemm(stream, yb3, p.fy_l1w, p.fy_l1b, nullptr, yb1, MY, cHID, cHID, cHID, cHID, cHID,
       1, 1, 0,0,0,0,0,0, 1.f, F_NT | F_RELU);
  gemm(stream, yb1, p.fy_l2w, p.fy_l2b, yb3, yb4, MY, cHID, cHID, cHID, cHID, cHID,
       1, 1, 0,0,0,0,0,0, 1.f, F_NT);
  ln_kernel<<<MY, 256, 0, stream>>>(yb4, p.n3y_g, p.n3y_b, cHID, 0);    // y = yb4

  // ---- trilinear attention pooling ----
  scale_cols_kernel<<<eg(SZX), 256, 0, stream>>>(bufC, p.att_w, bufA, SZX);
  gemm(stream, bufA, yb4, p.att_b, nullptr, bufS, cLV, cLQ, cHID, cHID, cHID, cLQ,
       cB, 1, XH, 0, YH, 0, (long long)cLV*cLQ, 0, 1.f, F_NT | F_SBIAS);
  softmax_kernel<<<dim3(cLV, cB), 256, 0, stream>>>(bufS, mq, mv, cLV, cLQ, 1, cLQ, cLV);
  gemm(stream, bufS, yb4, nullptr, nullptr, bufB, cLV, cHID, cLQ, cLQ, cHID, cHID,
       cB, 1, (long long)cLV*cLQ, 0, YH, 0, XH, 0, 1.f, 0);
  fuse_kernel<<<eg(SZX), 256, 0, stream>>>(bufC, bufB, p.alpha, bufA, SZX);

  // ---- MLP decoder ----
  gemm(stream, bufA, p.fc1w, p.fc1b, nullptr, bufB, MX, 640, cHID, cHID, cHID, 640,
       1, 1, 0,0,0,0,0,0, 1.f, F_NT | F_RELU);
  ln_kernel<<<MX, 256, 0, stream>>>(bufB, p.ln1g, p.ln1b, 640, 1);
  gemm(stream, bufB, p.fc2w, p.fc2b, nullptr, bufD, MX, 320, 640, 640, 640, 320,
       1, 1, 0,0,0,0,0,0, 1.f, F_NT | F_RELU);
  ln_kernel<<<MX, 256, 0, stream>>>(bufD, p.ln2g, p.ln2b, 320, 1);
  fc3_kernel<<<eg((long long)cB * (cLV - 2) * 2), 256, 0, stream>>>(
      bufD, p.fc3w, p.fc3b, (float*)d_out);
}
